// Encoderlayer_16492674417049
// MI455X (gfx1250) — matrix operand order repaired and checked by reading, not compiled, not run on hardware
//
#include <hip/hip_runtime.h>
#include <hip/hip_bf16.h>

typedef _Float16 half_t;
typedef __attribute__((ext_vector_type(16))) _Float16 v16h;
typedef __attribute__((ext_vector_type(8)))  _Float16 v8h;
typedef __attribute__((ext_vector_type(8)))  float    v8f;

#define DEV static __device__ __forceinline__

// ---------------- constants ----------------
// B=8, C_IN=64, C=128, H=W=128 -> after stride-2 conv: 64x64 (N=4096 tokens)
// NH=8, HD=16, Nk=1024 (32x32 after SR path)
static constexpr int BB    = 8;
static constexpr int CIN   = 64;
static constexpr int CC    = 128;
static constexpr int NTOK  = 4096;    // 64*64
static constexpr int NKV   = 1024;    // 32*32
static constexpr int MT    = BB * NTOK;   // 32768
static constexpr int MK    = BB * NKV;    // 8192
static constexpr int K1    = CIN * 9;     // 576  (conv1 im2col K)
static constexpr int K2    = CC  * 9;     // 1152 (ds conv im2col K)

// ---------------- WMMA helpers ----------------
DEV v8f zero8() { v8f z; for (int i = 0; i < 8; ++i) z[i] = 0.f; return z; }

DEV v8h zero8h() { v8h z; for (int i = 0; i < 8; ++i) z[i] = (half_t)0.f; return z; }

// A-fragment (16x32, f16) from row-major A tile, lda in elements.
// Lane layout: lanes 0-15 hold K 0..15; lanes 16-31 hold K 16..31.
DEV v16h load_a_frag(const half_t* Abase, int lda) {
    const int l  = threadIdx.x & 31;
    const int m  = l & 15;
    const int hi = l >> 4;
    const half_t* r = Abase + (size_t)m * lda + hi * 16;
    v8h lo  = *(const v8h*)r;          // K = hi*16 .. hi*16+7
    v8h hi8 = *(const v8h*)(r + 8);    // K = hi*16+8 .. hi*16+15
    return __builtin_shufflevector(lo, hi8, 0,1,2,3,4,5,6,7,8,9,10,11,12,13,14,15);
}

// A-fragment with K=16 real data, zero-padded to K=32 (for HD=16 Q tiles).
DEV v16h load_a_frag_k16(const half_t* Abase, int lda) {
    const int l  = threadIdx.x & 31;
    const int m  = l & 15;
    const int hi = l >> 4;
    const half_t* r = Abase + (size_t)m * lda;  // lanes 0-15: K = 0..15 (real)
    v8h lo = hi ? zero8h() : *(const v8h*)r;
    v8h z  = hi ? zero8h() : *(const v8h*)(r + 8); // lanes 16-31: K >= 16 -> zeros
    return __builtin_shufflevector(lo, z, 0,1,2,3,4,5,6,7,8,9,10,11,12,13,14,15);
}

// B-fragment (32x16, f16) where B = W^T; W is row-major [N][K], ldw in elements.
// Lane layout: lanes 0-15 -> K=0..15, lanes 16-31 -> K=16..31; column n = lane&15.
DEV v16h load_b_frag(const half_t* Wbase, int ldw) {
    const int l  = threadIdx.x & 31;
    const int n  = l & 15;
    const int kb = (l >> 4) * 16;
    return *(const v16h*)(Wbase + (size_t)n * ldw + kb);
}

DEV v8f wmma32(v16h a, v16h b, v8f c) {
    return __builtin_amdgcn_wmma_f32_16x16x32_f16(false, a, false, b, (short)0, c, false, false);
}

DEV float gelu_exact(float x) { return 0.5f * x * (1.0f + erff(x * 0.70710678118654752f)); }

// ---------------- generic GEMM: C[M,N] = A[M,K] * W[N,K]^T ----------------
// Block = 128 threads (4 waves), block tile 64x64, wave tile 32x32 (2x2 wmma).
// Requires: M%64==0, N%64==0, K%32==0, lda=ldw=K, ldo=N.
enum { EPI_F16_PLAIN = 0, EPI_F16_RELU = 1, EPI_F16_GELU = 2,
       EPI_F32_CONV1 = 3, EPI_F32_RES = 4, EPI_F16_VT = 5 };

template <int EPI>
__global__ void __launch_bounds__(128)
gemm_kernel(const half_t* __restrict__ A, const half_t* __restrict__ Wt,
            const float* __restrict__ bias, const float* __restrict__ rowscale,
            const float* __restrict__ res32, float* __restrict__ out32,
            half_t* __restrict__ out16, int M, int N, int K, float alpha) {
    const int wave = threadIdx.x >> 5;
    const int l    = threadIdx.x & 31;
    const int m0   = blockIdx.y * 64 + (wave & 1) * 32;
    const int n0   = blockIdx.x * 64 + (wave >> 1) * 32;

    v8f acc[2][2];
    acc[0][0] = zero8(); acc[0][1] = zero8(); acc[1][0] = zero8(); acc[1][1] = zero8();

    for (int kt = 0; kt < K; kt += 32) {
        v16h a0 = load_a_frag(A + (size_t)m0 * K + kt, K);
        v16h a1 = load_a_frag(A + (size_t)(m0 + 16) * K + kt, K);
        v16h b0 = load_b_frag(Wt + (size_t)n0 * K + kt, K);
        v16h b1 = load_b_frag(Wt + (size_t)(n0 + 16) * K + kt, K);
        acc[0][0] = wmma32(a0, b0, acc[0][0]);
        acc[0][1] = wmma32(a0, b1, acc[0][1]);
        acc[1][0] = wmma32(a1, b0, acc[1][0]);
        acc[1][1] = wmma32(a1, b1, acc[1][1]);
    }

    const int hi = l >> 4, nl = l & 15;
    for (int i = 0; i < 2; ++i) {
        for (int j = 0; j < 2; ++j) {
            for (int e = 0; e < 8; ++e) {
                const int m = m0 + i * 16 + e + 8 * hi;
                const int n = n0 + j * 16 + nl;
                float c = acc[i][j][e] * alpha;
                if (bias) c += bias[n];
                if (EPI == EPI_F16_PLAIN) {
                    out16[(size_t)m * N + n] = (half_t)c;
                } else if (EPI == EPI_F16_RELU) {
                    out16[(size_t)m * N + n] = (half_t)fmaxf(c, 0.f);
                } else if (EPI == EPI_F16_GELU) {
                    out16[(size_t)m * N + n] = (half_t)gelu_exact(c);
                } else if (EPI == EPI_F32_CONV1) {
                    out32[(size_t)m * N + n] = c * rowscale[m];
                } else if (EPI == EPI_F32_RES) {
                    out32[(size_t)m * N + n] = c + res32[(size_t)m * N + n];
                } else if (EPI == EPI_F16_VT) {
                    // write V transposed per head: vt[b][h][d][key], key-major
                    const int b  = m >> 10;       // m = b*1024 + key
                    const int mk = m & 1023;
                    const int h  = n >> 4;        // n = h*16 + d
                    const int d  = n & 15;
                    out16[(((size_t)b * 8 + h) * 16 + d) * 1024 + mk] = (half_t)c;
                }
            }
        }
    }
}

// ---------------- fused attention ----------------
// One block per (b, h, 32-query tile). Whole K (1024x16) and V^T (16x1024)
// staged in LDS; full 32x1024 f32 score slab + f16 P in LDS (CDNA5: 320KB/WGP).
static constexpr int SMEM_SCORES = 32 * 1024 * 4;   // 131072
static constexpr int SMEM_P      = 32 * 1024 * 2;   //  65536
static constexpr int SMEM_K      = 1024 * 16 * 2;   //  32768
static constexpr int SMEM_V      = 16 * 1024 * 2;   //  32768
static constexpr int SMEM_Q      = 32 * 16 * 2;     //   1024
static constexpr int SMEM_OB     = 4 * 16 * 16 * 4; //   4096
static constexpr int ATTN_SMEM   = SMEM_SCORES + SMEM_P + SMEM_K + SMEM_V + SMEM_Q + 256 + SMEM_OB;

__global__ void __launch_bounds__(128)
attn_kernel(const half_t* __restrict__ qf, const half_t* __restrict__ kf,
            const half_t* __restrict__ vt, half_t* __restrict__ of) {
    extern __shared__ unsigned char smem[];
    float*  s_s  = (float*)(smem);
    half_t* s_p  = (half_t*)(smem + SMEM_SCORES);
    half_t* s_k  = (half_t*)(smem + SMEM_SCORES + SMEM_P);
    half_t* s_v  = (half_t*)(smem + SMEM_SCORES + SMEM_P + SMEM_K);
    half_t* s_q  = (half_t*)(smem + SMEM_SCORES + SMEM_P + SMEM_K + SMEM_V);
    float*  s_l  = (float*)(smem + SMEM_SCORES + SMEM_P + SMEM_K + SMEM_V + SMEM_Q);
    float*  s_ob = (float*)(smem + SMEM_SCORES + SMEM_P + SMEM_K + SMEM_V + SMEM_Q + 256);

    const int tid   = threadIdx.x;
    const int blk   = blockIdx.x;       // ((b*8 + h)*128 + qt)
    const int qt    = blk & 127;
    const int h     = (blk >> 7) & 7;
    const int b     = blk >> 10;
    const int qbase = qt * 32;

    // phase 0: stage q (32x16), K (1024x16), V^T (16x1024) into LDS
    {
        const half_t* qsrc = qf + ((size_t)(b * NTOK + qbase)) * CC + h * 16;
        for (int i = tid; i < 64; i += 128) {
            const int r = i >> 1, p = (i & 1) * 8;
            *(v8h*)(s_q + r * 16 + p) = *(const v8h*)(qsrc + (size_t)r * CC + p);
        }
        const half_t* ksrc = kf + ((size_t)(b * NKV)) * CC + h * 16;
        for (int i = tid; i < 2048; i += 128) {
            const int r = i >> 1, p = (i & 1) * 8;
            *(v8h*)(s_k + r * 16 + p) = *(const v8h*)(ksrc + (size_t)r * CC + p);
        }
        const half_t* vsrc = vt + ((size_t)(b * 8 + h)) * 16 * 1024;
        for (int i = tid; i < 2048; i += 128) {
            *(v8h*)(s_v + i * 8) = *(const v8h*)(vsrc + i * 8);
        }
    }
    __syncthreads();

    // phase 1: S = q . K^T  (q pre-scaled by HD^-0.5 in its GEMM)
    {
        const int wave = tid >> 5, l = tid & 31, nl = l & 15, hi = l >> 4;
        v16h a0 = load_a_frag_k16(s_q, 16);            // query rows 0..15
        v16h a1 = load_a_frag_k16(s_q + 16 * 16, 16);  // query rows 16..31
        for (int t16 = 0; t16 < 16; ++t16) {
            const int nbase = wave * 256 + t16 * 16;
            v16h bfr = *(const v16h*)(s_k + (size_t)(nbase + nl) * 16); // K=0..15 real
            if (hi) { for (int z = 0; z < 16; ++z) bfr[z] = (half_t)0.f; } // K>=16 zeros
            v8f c0 = zero8(); c0 = wmma32(a0, bfr, c0);
            v8f c1 = zero8(); c1 = wmma32(a1, bfr, c1);
            for (int e = 0; e < 8; ++e) {
                s_s[(size_t)(e + 8 * hi) * 1024 + nbase + nl]      = c0[e];
                s_s[(size_t)(16 + e + 8 * hi) * 1024 + nbase + nl] = c1[e];
            }
        }
    }
    __syncthreads();

    // phase 2: softmax rows (4 lanes per row), P = exp(S - max) in f16, l = row sum
    {
        const int r = tid >> 2, q4 = tid & 3;
        const float* row = s_s + (size_t)r * 1024;
        float mx = -1e30f;
        for (int kk = q4; kk < 1024; kk += 4) mx = fmaxf(mx, row[kk]);
        mx = fmaxf(mx, __shfl_xor(mx, 1, 4));
        mx = fmaxf(mx, __shfl_xor(mx, 2, 4));
        float lsum = 0.f;
        half_t* prow = s_p + (size_t)r * 1024;
        for (int kk = q4; kk < 1024; kk += 4) {
            const float e = __expf(row[kk] - mx);
            prow[kk] = (half_t)e;
            lsum += e;
        }
        lsum += __shfl_xor(lsum, 1, 4);
        lsum += __shfl_xor(lsum, 2, 4);
        if (q4 == 0) s_l[r] = lsum;
    }
    __syncthreads();

    // phase 3: O = P . V  (K = 1024, split across waves: 2 m-tiles x 2 k-halves)
    {
        const int wave = tid >> 5, l = tid & 31, nl = l & 15, hi = l >> 4;
        const int mtile = wave & 1, khalf = wave >> 1;
        v8f acc = zero8();
        const half_t* Abase = s_p + (size_t)mtile * 16 * 1024;
        for (int ks = khalf * 512; ks < khalf * 512 + 512; ks += 32) {
            v16h a   = load_a_frag(Abase + ks, 1024);
            v16h bfr = *(const v16h*)(s_v + (size_t)nl * 1024 + ks + hi * 16);
            acc = wmma32(a, bfr, acc);
        }
        float* ob = s_ob + (mtile * 2 + khalf) * 256;
        for (int e = 0; e < 8; ++e) ob[(e + 8 * hi) * 16 + nl] = acc[e];
    }
    __syncthreads();

    // phase 4: combine k-halves, normalize, scatter heads into o tokens
    for (int e = tid; e < 512; e += 128) {
        const int r = e >> 4, d = e & 15;
        const int mtile = r >> 4, rr = r & 15;
        const float v0 = s_ob[(mtile * 2 + 0) * 256 + rr * 16 + d];
        const float v1 = s_ob[(mtile * 2 + 1) * 256 + rr * 16 + d];
        const float o  = (v0 + v1) / s_l[r];
        of[((size_t)(b * NTOK + qbase + r)) * CC + h * 16 + d] = (half_t)o;
    }
}

// ---------------- elementwise / staging kernels ----------------
__global__ void f2h_kernel(const float* __restrict__ in, half_t* __restrict__ out, int n) {
    const int i = blockIdx.x * 256 + threadIdx.x;
    if (i < n) out[i] = (half_t)in[i];
}

// partial-conv mask: um = 3x3 stride-2 pad-1 box sum of mask; ratio & update_mask
__global__ void maskratio_kernel(const float* __restrict__ mask,
                                 float* __restrict__ ratio, float* __restrict__ um_out) {
    const int idx = blockIdx.x * 256 + threadIdx.x;
    if (idx >= BB * NTOK) return;
    const int b = idx >> 12, p = idx & 4095, oh = p >> 6, ow = p & 63;
    float um = 0.f;
    for (int kh = 0; kh < 3; ++kh) {
        for (int kw = 0; kw < 3; ++kw) {
            const int iy = 2 * oh - 1 + kh, ix = 2 * ow - 1 + kw;
            if ((unsigned)iy < 128u && (unsigned)ix < 128u)
                um += mask[((size_t)b * 128 + iy) * 128 + ix];
        }
    }
    const float cl = fminf(fmaxf(um, 0.f), 1.f);
    ratio[idx]  = (9.0f / (um + 1e-8f)) * cl;
    um_out[idx] = cl;
}

// im2col for conv1: x[8,64,128,128] f32 -> A1[32768, 576] f16 (stride 2, pad 1)
__global__ void im2col1_kernel(const float* __restrict__ x, half_t* __restrict__ a1) {
    const int idx = blockIdx.x * 256 + threadIdx.x;
    if (idx >= MT * K1) return;
    const int m = idx / K1, kk = idx % K1;
    const int b = m >> 12, p = m & 4095, oh = p >> 6, ow = p & 63;
    const int ci = kk / 9, r = kk % 9, kh = r / 3, kwp = r % 3;
    const int iy = 2 * oh - 1 + kh, ix = 2 * ow - 1 + kwp;
    float v = 0.f;
    if ((unsigned)iy < 128u && (unsigned)ix < 128u)
        v = x[(((size_t)b * CIN + ci) * 128 + iy) * 128 + ix];
    a1[idx] = (half_t)v;
}

// im2col for SR conv: y tokens [b,4096,128] f16 (as 64x64 image) -> A2[8192, 1152]
__global__ void im2col2_kernel(const half_t* __restrict__ y, half_t* __restrict__ a2) {
    const int idx = blockIdx.x * 256 + threadIdx.x;
    if (idx >= MK * K2) return;
    const int m = idx / K2, kk = idx % K2;
    const int b = m >> 10, p = m & 1023, oh = p >> 5, ow = p & 31;
    const int c = kk / 9, r = kk % 9, kh = r / 3, kwp = r % 3;
    const int iy = 2 * oh - 1 + kh, ix = 2 * ow - 1 + kwp;
    half_t v = (half_t)0.f;
    if ((unsigned)iy < 64u && (unsigned)ix < 64u)
        v = y[((size_t)(b << 12) + iy * 64 + ix) * CC + c];
    a2[idx] = v;
}

// LayerNorm over C=128, one wave per token; f32 in -> f16 out
__global__ void __launch_bounds__(128)
ln_kernel(const float* __restrict__ t, const float* __restrict__ g,
          const float* __restrict__ bt, half_t* __restrict__ y) {
    const int tok = blockIdx.x * 4 + (threadIdx.x >> 5);
    const int l   = threadIdx.x & 31;
    const float* row = t + (size_t)tok * CC;
    const float4 v = *(const float4*)(row + l * 4);
    float s  = v.x + v.y + v.z + v.w;
    float s2 = v.x * v.x + v.y * v.y + v.z * v.z + v.w * v.w;
    for (int m = 1; m < 32; m <<= 1) { s += __shfl_xor(s, m, 32); s2 += __shfl_xor(s2, m, 32); }
    const float mu  = s * (1.f / 128.f);
    const float var = s2 * (1.f / 128.f) - mu * mu;
    const float rs  = rsqrtf(var + 1e-5f);
    const int c = l * 4;
    half_t* yr = y + (size_t)tok * CC;
    yr[c + 0] = (half_t)((v.x - mu) * rs * g[c + 0] + bt[c + 0]);
    yr[c + 1] = (half_t)((v.y - mu) * rs * g[c + 1] + bt[c + 1]);
    yr[c + 2] = (half_t)((v.z - mu) * rs * g[c + 2] + bt[c + 2]);
    yr[c + 3] = (half_t)((v.w - mu) * rs * g[c + 3] + bt[c + 3]);
}

// token2feature: t[b,4096,128] f32 -> out[b,128,64,64]
__global__ void t2f_kernel(const float* __restrict__ t, float* __restrict__ out) {
    const int i = blockIdx.x * 256 + threadIdx.x;
    if (i >= BB * CC * NTOK) return;
    const int b = i >> 19, c = (i >> 12) & 127, p = i & 4095;
    out[i] = t[((size_t)(b << 12) + p) * CC + c];
}

// ---------------- launch ----------------
extern "C" void kernel_launch(void* const* d_in, const int* in_sizes, int n_in,
                              void* d_out, int out_size, void* d_ws, size_t ws_size,
                              hipStream_t stream) {
    (void)in_sizes; (void)n_in; (void)out_size; (void)ws_size;

    const float* x      = (const float*)d_in[0];
    const float* mask   = (const float*)d_in[1];
    const float* conv_w = (const float*)d_in[2];
    const float* conv_b = (const float*)d_in[3];
    const float* n1w    = (const float*)d_in[4];
    const float* n1b    = (const float*)d_in[5];
    const float* qw     = (const float*)d_in[6];
    const float* kw     = (const float*)d_in[7];
    const float* vw     = (const float*)d_in[8];
    const float* pw     = (const float*)d_in[9];
    const float* pb     = (const float*)d_in[10];
    const float* dsw    = (const float*)d_in[11];
    const float* dsb    = (const float*)d_in[12];
    const float* c1w    = (const float*)d_in[13];
    const float* c1b    = (const float*)d_in[14];
    const float* n2w    = (const float*)d_in[15];
    const float* n2b    = (const float*)d_in[16];
    const float* f1w    = (const float*)d_in[17];
    const float* f1b    = (const float*)d_in[18];
    const float* f2w    = (const float*)d_in[19];
    const float* f2b    = (const float*)d_in[20];

    float* out       = (float*)d_out;
    float* umask_out = out + (size_t)BB * CC * NTOK;

    char*  ws  = (char*)d_ws;
    size_t off = 0;
    auto alloc = [&](size_t bytes) -> char* {
        char* p = ws + off;
        off = (off + bytes + 255) & ~(size_t)255;
        return p;
    };

    half_t* wf_conv = (half_t*)alloc((size_t)CC * K1 * 2);            // [128,576]
    half_t* wf_q    = (half_t*)alloc((size_t)2 * CC * CC * 2);
    half_t* wf_k    = (half_t*)alloc((size_t)2 * CC * CC * 2);
    half_t* wf_v    = (half_t*)alloc((size_t)2 * CC * CC * 2);
    half_t* wf_p    = (half_t*)alloc((size_t)2 * CC * CC * 2);
    half_t* wf_ds   = (half_t*)alloc((size_t)2 * CC * K2 * 2);        // [2,128,1152]
    half_t* wf_c1   = (half_t*)alloc((size_t)2 * CC * CC * 2);
    half_t* wf_f1   = (half_t*)alloc((size_t)2 * 512 * CC * 2);
    half_t* wf_f2   = (half_t*)alloc((size_t)2 * CC * 512 * 2);
    float*  ratio   = (float*) alloc((size_t)MT * 4);
    half_t* A1      = (half_t*)alloc((size_t)MT * K1 * 2);
    float*  t       = (float*) alloc((size_t)MT * CC * 4);            // residual stream
    half_t* y       = (half_t*)alloc((size_t)MT * CC * 2);            // LN1 out
    half_t* A2      = (half_t*)alloc((size_t)MK * K2 * 2);
    half_t* dso     = (half_t*)alloc((size_t)MK * CC * 2);
    half_t* kv      = (half_t*)alloc((size_t)MK * CC * 2);
    half_t* qf      = (half_t*)alloc((size_t)MT * CC * 2);
    half_t* kfb     = (half_t*)alloc((size_t)MK * CC * 2);
    half_t* vtb     = (half_t*)alloc((size_t)BB * 8 * 16 * 1024 * 2); // V^T [b,h,d,key]
    half_t* ofb     = (half_t*)alloc((size_t)MT * CC * 2);
    half_t* zf      = (half_t*)alloc((size_t)MT * CC * 2);            // LN2 out
    half_t* h1      = (half_t*)alloc((size_t)MT * 512 * 2);

    auto cvt = [&](const float* src, half_t* dst, int n) {
        f2h_kernel<<<(n + 255) / 256, 256, 0, stream>>>(src, dst, n);
    };
    cvt(conv_w, wf_conv, CC * K1);
    cvt(qw,  wf_q,  2 * CC * CC);
    cvt(kw,  wf_k,  2 * CC * CC);
    cvt(vw,  wf_v,  2 * CC * CC);
    cvt(pw,  wf_p,  2 * CC * CC);
    cvt(dsw, wf_ds, 2 * CC * K2);
    cvt(c1w, wf_c1, 2 * CC * CC);
    cvt(f1w, wf_f1, 2 * 512 * CC);
    cvt(f2w, wf_f2, 2 * CC * 512);

    maskratio_kernel<<<(MT + 255) / 256, 256, 0, stream>>>(mask, ratio, umask_out);
    im2col1_kernel<<<(MT * K1) / 256, 256, 0, stream>>>(x, A1);

    // partial-conv downsample: t = (im2col(x) @ conv_w^T + b) * ratio   [32768,128]
    gemm_kernel<EPI_F32_CONV1><<<dim3(CC / 64, MT / 64), 128, 0, stream>>>(
        A1, wf_conv, conv_b, ratio, nullptr, t, nullptr, MT, CC, K1, 1.0f);

    for (int i = 0; i < 2; ++i) {
        const half_t* qwi  = wf_q  + (size_t)i * CC * CC;
        const half_t* kwi  = wf_k  + (size_t)i * CC * CC;
        const half_t* vwi  = wf_v  + (size_t)i * CC * CC;
        const half_t* pwi  = wf_p  + (size_t)i * CC * CC;
        const half_t* dswi = wf_ds + (size_t)i * CC * K2;
        const half_t* c1wi = wf_c1 + (size_t)i * CC * CC;
        const half_t* f1wi = wf_f1 + (size_t)i * 512 * CC;
        const half_t* f2wi = wf_f2 + (size_t)i * CC * 512;

        // LN1: t -> y (f16)
        ln_kernel<<<MT / 4, 128, 0, stream>>>(t, n1w + i * CC, n1b + i * CC, y);

        // spatial-reduction path: ds conv (3x3 s2) + 1x1 conv + ReLU -> kv tokens
        im2col2_kernel<<<(MK * K2) / 256, 256, 0, stream>>>(y, A2);
        gemm_kernel<EPI_F16_PLAIN><<<dim3(CC / 64, MK / 64), 128, 0, stream>>>(
            A2, dswi, dsb + i * CC, nullptr, nullptr, nullptr, dso, MK, CC, K2, 1.0f);
        gemm_kernel<EPI_F16_RELU><<<dim3(CC / 64, MK / 64), 128, 0, stream>>>(
            dso, c1wi, c1b + i * CC, nullptr, nullptr, nullptr, kv, MK, CC, CC, 1.0f);

        // q (scaled by HD^-0.5), k, v^T
        gemm_kernel<EPI_F16_PLAIN><<<dim3(CC / 64, MT / 64), 128, 0, stream>>>(
            y, qwi, nullptr, nullptr, nullptr, nullptr, qf, MT, CC, CC, 0.25f);
        gemm_kernel<EPI_F16_PLAIN><<<dim3(CC / 64, MK / 64), 128, 0, stream>>>(
            kv, kwi, nullptr, nullptr, nullptr, nullptr, kfb, MK, CC, CC, 1.0f);
        gemm_kernel<EPI_F16_VT><<<dim3(CC / 64, MK / 64), 128, 0, stream>>>(
            kv, vwi, nullptr, nullptr, nullptr, nullptr, vtb, MK, CC, CC, 1.0f);

        // fused attention: 8192 blocks, whole K/V per (b,h) resident in LDS
        attn_kernel<<<BB * 8 * (NTOK / 32), 128, ATTN_SMEM, stream>>>(qf, kfb, vtb, ofb);

        // proj + residual (in-place on t)
        gemm_kernel<EPI_F32_RES><<<dim3(CC / 64, MT / 64), 128, 0, stream>>>(
            ofb, pwi, pb + i * CC, nullptr, t, t, nullptr, MT, CC, CC, 1.0f);

        // LN2 -> MLP (GELU) + residual (in-place on t)
        ln_kernel<<<MT / 4, 128, 0, stream>>>(t, n2w + i * CC, n2b + i * CC, zf);
        gemm_kernel<EPI_F16_GELU><<<dim3(512 / 64, MT / 64), 128, 0, stream>>>(
            zf, f1wi, f1b + i * 512, nullptr, nullptr, nullptr, h1, MT, 512, CC, 1.0f);
        gemm_kernel<EPI_F32_RES><<<dim3(CC / 64, MT / 64), 128, 0, stream>>>(
            h1, f2wi, f2b + i * CC, nullptr, t, t, nullptr, MT, CC, 512, 1.0f);
    }

    // token2feature: t -> out [8,128,64,64]
    t2f_kernel<<<(BB * CC * NTOK) / 256, 256, 0, stream>>>(t, out);
}
